// H2_EdgeAtt_88098369176183
// MI455X (gfx1250) — compile-verified, matrix operand order checked by reading, not run on hardware
//
#include <hip/hip_runtime.h>
#include <hip/hip_bf16.h>

// ---------------------------------------------------------------------------
// GNN forward for MI455X (gfx1250, wave32).
//  - All dense GEMMs (K in {128,256,384}, N in {32,128}) via
//    v_wmma_f32_16x16x32_bf16, LDS-staged bf16 tiles, fused epilogues.
//  - Edge softmax / segment ops via L2 atomics (feature matrices are
//    ~51MB => L2-resident on the 192MB L2, so random gathers are cheap).
// ---------------------------------------------------------------------------

#define N_A_   100000
#define HID_   128
#define E_     800000
#define N_C2_  20000
#define EI_    160000
#define NB_    128

typedef __attribute__((ext_vector_type(16))) __bf16       v16bf;
typedef __attribute__((ext_vector_type(8)))  float        v8f;
typedef __attribute__((ext_vector_type(4)))  unsigned int v4u;

struct FragBits { v4u lo, hi; };  // 32 bytes == v16bf

__device__ __forceinline__ unsigned short f32_to_bf16(float f) {
  unsigned u = __float_as_uint(f);
  u += 0x7fffu + ((u >> 16) & 1u);   // round-to-nearest-even
  return (unsigned short)(u >> 16);
}

// order-preserving encode of signed float into unsigned for atomicMax
__device__ __forceinline__ unsigned encf(float f) {
  unsigned u = __float_as_uint(f);
  return (u & 0x80000000u) ? ~u : (u | 0x80000000u);
}
__device__ __forceinline__ float decf(unsigned k) {
  return __uint_as_float((k & 0x80000000u) ? (k ^ 0x80000000u) : ~k);
}

// ---------------------------------------------------------------------------
// WMMA GEMM: C[M,NCOLS] = act(Aload(A)[M,K] @ W[K,NCOLS] + bias)
//   mode 0: A dense fp32
//   mode 1: A row e = concat(pca[gsrc[e]], pca[gdst[e]])  (K==128, ATT==64)
//   mode 2: A = relu((x - mu)*rstd*gamma + beta)          (fused BatchNorm)
// Block: 256 threads = 8 waves; tile 128(M) x NCOLS; K chunked by 64 via LDS.
// ---------------------------------------------------------------------------
template<int NCOLS>
__global__ __launch_bounds__(256)
void wmma_gemm(const float* __restrict__ A, const float* __restrict__ W,
               const float* __restrict__ bias, float* __restrict__ C,
               int M, int K, int ldc, int col0, int mode, int act,
               const int* __restrict__ gsrc, const int* __restrict__ gdst,
               const float* __restrict__ pca,
               const float* __restrict__ mu, const float* __restrict__ rstd,
               const float* __restrict__ gam, const float* __restrict__ bet) {
  constexpr int NT = NCOLS / 16;
  constexpr int KC = 64, LD = 72;            // +8 shorts pad: bank-skewed rows
  __shared__ __align__(16) unsigned short sA[128 * LD];
  __shared__ __align__(16) unsigned short sB[NCOLS * LD];

  const int tid  = threadIdx.x;
  const int lane = tid & 31, wv = tid >> 5;
  const int half = lane >> 4, r = lane & 15;
  const int m0   = blockIdx.x * 128;

  v8f acc[NT];
  for (int i = 0; i < NT; ++i)
    for (int j = 0; j < 8; ++j) acc[i][j] = 0.f;

  for (int k0 = 0; k0 < K; k0 += KC) {
    // stage A chunk (fp32 -> bf16, with fused gather / BN+ReLU)
    for (int idx = tid; idx < 128 * KC; idx += 256) {
      int row = idx >> 6, col = idx & 63;
      int m = m0 + row, k = k0 + col;
      float v = 0.f;
      if (m < M) {
        if (mode == 1) {
          int node = (k < 64) ? gsrc[m] : gdst[m];
          v = pca[(size_t)node * 64 + (k & 63)];
        } else {
          v = A[(size_t)m * K + k];
          if (mode == 2) {
            v = (v - mu[k]) * rstd[k] * gam[k] + bet[k];
            v = fmaxf(v, 0.f);
          }
        }
      }
      sA[row * LD + col] = f32_to_bf16(v);
    }
    // stage W chunk transposed: sB[n][c] = W[k0+c][n]
    for (int idx = tid; idx < NCOLS * KC; idx += 256) {
      int n = idx >> 6, c = idx & 63;
      sB[n * LD + c] = f32_to_bf16(W[(size_t)(k0 + c) * NCOLS + n]);
    }
    __syncthreads();

#pragma unroll
    for (int ks = 0; ks < KC; ks += 32) {
      // A fragment: two contiguous 16B runs per lane (16-bit A layout)
      FragBits fa;
      const unsigned short* pa = &sA[(wv * 16 + r) * LD + ks + half * 8];
      fa.lo = *(const v4u*)pa;
      fa.hi = *(const v4u*)(pa + 16);
      v16bf av = __builtin_bit_cast(v16bf, fa);
#pragma unroll
      for (int nt = 0; nt < NT; ++nt) {
        FragBits fb;
        const unsigned short* pb = &sB[(nt * 16 + r) * LD + ks + half * 8];
        fb.lo = *(const v4u*)pb;
        fb.hi = *(const v4u*)(pb + 16);
        v16bf bv = __builtin_bit_cast(v16bf, fb);
        acc[nt] = __builtin_amdgcn_wmma_f32_16x16x32_bf16(
            false, av, false, bv, (short)0, acc[nt], false, false);
      }
    }
    __syncthreads();
  }

  // epilogue: C/D layout — VGPR j: M = j + 8*half, lanes%16 = N
#pragma unroll
  for (int nt = 0; nt < NT; ++nt) {
    int n = nt * 16 + r;
    float bv = bias ? bias[n] : 0.f;
#pragma unroll
    for (int j = 0; j < 8; ++j) {
      int m = m0 + wv * 16 + half * 8 + j;
      if (m < M) {
        float v = acc[nt][j] + bv;
        if (act == 1) v = fmaxf(v, 0.f);
        C[(size_t)m * ldc + col0 + n] = v;
      }
    }
  }
}

// ---------------------------------------------------------------------------
__global__ void fill_u32_kernel(unsigned* p, unsigned v, int n) {
  int i = blockIdx.x * blockDim.x + threadIdx.x;
  if (i < n) p[i] = v;
}

__global__ void el_er_kernel(const float* __restrict__ f,
                             const float* __restrict__ al,
                             const float* __restrict__ ar,
                             float* __restrict__ el, float* __restrict__ er,
                             int n_nodes) {
  int i = blockIdx.x * blockDim.x + threadIdx.x;
  if (i >= n_nodes * 4) return;
  int h = i & 3, node = i >> 2;
  const float* row = f + (size_t)node * 128 + h * 32;
  float sl = 0.f, sr = 0.f;
#pragma unroll
  for (int d = 0; d < 32; ++d) {
    float v = row[d];
    sl += v * al[h * 32 + d];
    sr += v * ar[h * 32 + d];
  }
  el[i] = sl; er[i] = sr;
}

__global__ void gat_edge_score_kernel(const int* __restrict__ src,
                                      const int* __restrict__ dst,
                                      const float* __restrict__ el,
                                      const float* __restrict__ er,
                                      float* __restrict__ sc,
                                      unsigned* __restrict__ mx, int ne) {
  int i = blockIdx.x * blockDim.x + threadIdx.x;
  if (i >= ne * 4) return;
  int h = i & 3, e = i >> 2;
  float v = el[src[e] * 4 + h] + er[dst[e] * 4 + h];
  v = (v > 0.f) ? v : 0.2f * v;                 // leaky 0.2
  sc[i] = v;
  atomicMax(&mx[dst[e] * 4 + h], encf(v));
}

__global__ void edge_exp_sum_kernel(const int* __restrict__ dst,
                                    float* __restrict__ sc,
                                    const unsigned* __restrict__ mx,
                                    float* __restrict__ ssum, int ne) {
  int i = blockIdx.x * blockDim.x + threadIdx.x;
  if (i >= ne * 4) return;
  int h = i & 3, e = i >> 2;
  int d = dst[e];
  float ex = __expf(sc[i] - decf(mx[d * 4 + h]));
  sc[i] = ex;
  atomicAdd(&ssum[d * 4 + h], ex);
}

__global__ void gat_accum_kernel(const int* __restrict__ src,
                                 const int* __restrict__ dst,
                                 const float* __restrict__ sc,
                                 const float* __restrict__ ssum,
                                 const float* __restrict__ f,
                                 float* __restrict__ hacc, int ne) {
  int gid = blockIdx.x * blockDim.x + threadIdx.x;  // ne*32 threads
  if (gid >= ne * 32) return;
  int e = gid >> 5, q = (gid & 31) * 4, h = q >> 5;
  int s = src[e], d = dst[e];
  float a = sc[e * 4 + h] / ssum[d * 4 + h];
  float4 fv = *(const float4*)(f + (size_t)s * 128 + q);
  float* p = hacc + (size_t)d * 128 + q;
  atomicAdd(p + 0, fv.x * a); atomicAdd(p + 1, fv.y * a);
  atomicAdd(p + 2, fv.z * a); atomicAdd(p + 3, fv.w * a);
}

__global__ void bias_relu_kernel(const float* __restrict__ x,
                                 const float* __restrict__ b,
                                 float* __restrict__ y, int n) {
  int i = blockIdx.x * blockDim.x + threadIdx.x;
  if (i >= n) return;
  y[i] = fmaxf(x[i] + b[i & 127], 0.f);
}

__global__ void scatter_add_kernel(const int* __restrict__ src,
                                   const int* __restrict__ dst,
                                   const float* __restrict__ h,
                                   float* __restrict__ agg, int ne) {
  int gid = blockIdx.x * blockDim.x + threadIdx.x;
  if (gid >= ne * 32) return;
  int e = gid >> 5, q = (gid & 31) * 4;
  float4 v = *(const float4*)(h + (size_t)src[e] * 128 + q);
  float* p = agg + (size_t)dst[e] * 128 + q;
  atomicAdd(p + 0, v.x); atomicAdd(p + 1, v.y);
  atomicAdd(p + 2, v.z); atomicAdd(p + 3, v.w);
}

__global__ void gin_combine_kernel(const float* __restrict__ h,
                                   const float* __restrict__ agg,
                                   const float* __restrict__ epsp, int t,
                                   float* __restrict__ x, int n) {
  int i = blockIdx.x * blockDim.x + threadIdx.x;
  if (i >= n) return;
  x[i] = (1.f + epsp[t]) * h[i] + agg[i];
}

__global__ void bn_stats_kernel(const float* __restrict__ x,
                                float* __restrict__ sums, int n_rows) {
  __shared__ float ls[256], lq[256];
  int tid = threadIdx.x, c = tid & 127, g = tid >> 7;
  int r0 = blockIdx.x * 256;
  int rend = min(r0 + 256, n_rows);
  float s = 0.f, q = 0.f;
  for (int r = r0 + g; r < rend; r += 2) {
    float v = x[(size_t)r * 128 + c];
    s += v; q += v * v;
  }
  ls[tid] = s; lq[tid] = q;
  __syncthreads();
  if (g == 0) {
    atomicAdd(&sums[c],       ls[c] + ls[c + 128]);
    atomicAdd(&sums[128 + c], lq[c] + lq[c + 128]);
  }
}

__global__ void bn_finalize_kernel(const float* __restrict__ sums,
                                   float* __restrict__ murstd, float inv_n) {
  int c = threadIdx.x;
  if (c >= 128) return;
  float mu  = sums[c] * inv_n;
  float var = sums[128 + c] * inv_n - mu * mu;
  murstd[c] = mu;
  murstd[128 + c] = rsqrtf(var + 1e-5f);
}

__global__ void segmax_kernel(const float* __restrict__ hA,
                              const int* __restrict__ g1,
                              unsigned* __restrict__ hC, int n) {
  int gid = blockIdx.x * blockDim.x + threadIdx.x;  // n*64
  if (gid >= n * 64) return;
  int node = gid >> 6, q = (gid & 63) * 4;
  int c2 = g1[node];
  float4 v = *(const float4*)(hA + (size_t)node * 256 + q);
  unsigned* p = hC + (size_t)c2 * 256 + q;
  // hA >= 0 (ReLU output): plain uint order == float order, init 0
  atomicMax(p + 0, __float_as_uint(fmaxf(v.x, 0.f)));
  atomicMax(p + 1, __float_as_uint(fmaxf(v.y, 0.f)));
  atomicMax(p + 2, __float_as_uint(fmaxf(v.z, 0.f)));
  atomicMax(p + 3, __float_as_uint(fmaxf(v.w, 0.f)));
}

__global__ void att_edge_score_kernel(const float* __restrict__ tatt,
                                      const float* __restrict__ W2,
                                      const float* __restrict__ b2,
                                      const int* __restrict__ dst,
                                      float* __restrict__ sc,
                                      unsigned* __restrict__ mx, int ne) {
  int i = blockIdx.x * blockDim.x + threadIdx.x;
  if (i >= ne * 4) return;
  int h = i & 3, e = i >> 2;
  const float* row = tatt + (size_t)e * 128;
  float s = b2[h];
#pragma unroll 8
  for (int k = 0; k < 128; ++k) s += row[k] * W2[k * 4 + h];
  s = (s > 0.f) ? s : 0.01f * s;                 // leaky 0.01
  sc[i] = s;
  atomicMax(&mx[dst[e] * 4 + h], encf(s));
}

__global__ void att_accum_kernel(const int* __restrict__ src,
                                 const int* __restrict__ dst,
                                 const float* __restrict__ sc,
                                 const float* __restrict__ ssum,
                                 const float* __restrict__ hm,
                                 float* __restrict__ hC2, int ne) {
  int gid = blockIdx.x * blockDim.x + threadIdx.x;  // ne*32
  if (gid >= ne * 32) return;
  int e = gid >> 5, q = (gid & 31) * 4, h = q >> 5;
  float a = sc[e * 4 + h] / ssum[dst[e] * 4 + h];
  float4 v = *(const float4*)(hm + (size_t)src[e] * 32 + (q & 31));
  float* p = hC2 + (size_t)dst[e] * 128 + q;
  atomicAdd(p + 0, v.x * a); atomicAdd(p + 1, v.y * a);
  atomicAdd(p + 2, v.z * a); atomicAdd(p + 3, v.w * a);
}

__global__ void count_kernel(const int* __restrict__ batch,
                             float* __restrict__ cnt, int n) {
  int i = blockIdx.x * blockDim.x + threadIdx.x;
  if (i < n) atomicAdd(&cnt[batch[i]], 1.f);
}

__global__ void readout_sum_kernel(const float* __restrict__ x,
                                   const int* __restrict__ batch,
                                   float* __restrict__ sum, int n,
                                   int width, int do_relu) {
  int per = width >> 2;
  int gid = blockIdx.x * blockDim.x + threadIdx.x;
  if (gid >= n * per) return;
  int node = gid / per, q = (gid % per) * 4;
  float4 v = *(const float4*)(x + (size_t)node * width + q);
  if (do_relu) {
    v.x = fmaxf(v.x, 0.f); v.y = fmaxf(v.y, 0.f);
    v.z = fmaxf(v.z, 0.f); v.w = fmaxf(v.w, 0.f);
  }
  float* p = sum + (size_t)batch[node] * width + q;
  atomicAdd(p + 0, v.x); atomicAdd(p + 1, v.y);
  atomicAdd(p + 2, v.z); atomicAdd(p + 3, v.w);
}

__global__ void hcat_kernel(const float* __restrict__ h1s,
                            const float* __restrict__ h2s,
                            const float* __restrict__ cA,
                            const float* __restrict__ cC,
                            float* __restrict__ hcat) {
  int i = blockIdx.x * blockDim.x + threadIdx.x;
  if (i >= NB_ * 384) return;
  int b = i / 384, c = i % 384;
  hcat[i] = (c < 256) ? h1s[b * 256 + c] / cA[b]
                      : h2s[b * 128 + (c - 256)] / cC[b];
}

__global__ void final_out_kernel(const float* __restrict__ f2,
                                 const float* __restrict__ W3,
                                 const float* __restrict__ b3,
                                 float* __restrict__ out) {
  int b = threadIdx.x;
  if (b >= NB_) return;
  float s = b3[0];
#pragma unroll 8
  for (int k = 0; k < 128; ++k) s += f2[b * 128 + k] * W3[k];
  out[b] = s;
}

// ---------------------------------------------------------------------------
extern "C" void kernel_launch(void* const* d_in, const int* in_sizes, int n_in,
                              void* d_out, int out_size, void* d_ws,
                              size_t ws_size, hipStream_t stream) {
  const float* feats  = (const float*)d_in[0];
  const float* pca    = (const float*)d_in[1];
  const float* gat_W  = (const float*)d_in[2];
  const float* gat_al = (const float*)d_in[3];
  const float* gat_ar = (const float*)d_in[4];
  const float* gat_b  = (const float*)d_in[5];
  const float* gin_W1 = (const float*)d_in[6];
  const float* gin_b1 = (const float*)d_in[7];
  const float* gin_g  = (const float*)d_in[8];
  const float* gin_be = (const float*)d_in[9];
  const float* gin_W2 = (const float*)d_in[10];
  const float* gin_b2 = (const float*)d_in[11];
  const float* gin_ep = (const float*)d_in[12];
  const float* h2_W1  = (const float*)d_in[13];
  const float* h2_b1  = (const float*)d_in[14];
  const float* h2_W2  = (const float*)d_in[15];
  const float* h2_b2  = (const float*)d_in[16];
  const float* att_W1 = (const float*)d_in[17];
  const float* att_b1 = (const float*)d_in[18];
  const float* att_W2 = (const float*)d_in[19];
  const float* att_b2 = (const float*)d_in[20];
  const float* out_W1 = (const float*)d_in[21];
  const float* out_b1 = (const float*)d_in[22];
  const float* out_W2 = (const float*)d_in[23];
  const float* out_b2 = (const float*)d_in[24];
  const float* out_W3 = (const float*)d_in[25];
  const float* out_b3 = (const float*)d_in[26];
  const int*   e_src  = (const int*)d_in[27];
  const int*   e_dst  = (const int*)d_in[28];
  const int*   g1d    = (const int*)d_in[29];
  const int*   i2s    = (const int*)d_in[30];
  const int*   i2d    = (const int*)d_in[31];
  const int*   bA     = (const int*)d_in[32];
  const int*   bC     = (const int*)d_in[33];

  // workspace layout (256B aligned slots)
  size_t off = 0;
  char* base = (char*)d_ws;
  auto take = [&](size_t nflt) -> float* {
    float* p = (float*)(base + off);
    off += ((nflt * 4 + 255) & ~(size_t)255);
    return p;
  };
  float* hA   = take((size_t)N_A_ * 256);
  float* fbuf = take((size_t)N_A_ * 128);   // f, then reused as agg
  float* hacc = take((size_t)N_A_ * 128);   // GAT accumulator, then x_in
  float* hgat = take((size_t)N_A_ * 128);
  float* xb   = take((size_t)N_A_ * 128);
  float* el   = take((size_t)N_A_ * 4);
  float* er   = take((size_t)N_A_ * 4);
  float* esc  = take((size_t)E_ * 4);
  float* mbuf = take((size_t)N_A_ * 4);
  float* sbuf = take((size_t)N_A_ * 4);
  float* bns  = take(256);
  float* bnm  = take(256);
  float* hC   = take((size_t)N_C2_ * 256);
  float* hmt  = take((size_t)N_C2_ * 128);
  float* hm   = take((size_t)N_C2_ * 32);
  float* tatt = take((size_t)EI_ * 128);
  float* asc  = take((size_t)EI_ * 4);
  float* mC   = take((size_t)N_C2_ * 4);
  float* sC   = take((size_t)N_C2_ * 4);
  float* hC2  = take((size_t)N_C2_ * 128);
  float* h1s  = take((size_t)NB_ * 256);
  float* h2s  = take((size_t)NB_ * 128);
  float* cA   = take(NB_);
  float* cC   = take(NB_);
  float* hcat = take((size_t)NB_ * 384);
  float* f1   = take((size_t)NB_ * 128);
  float* f2   = take((size_t)NB_ * 128);

  auto cdiv = [](long long a, long long b) { return (int)((a + b - 1) / b); };
  const unsigned ENC_NEG_INF = 0x007fffffu;   // encf(-inf)

  for (int t = 0; t < 2; ++t) {
    const int* src = e_src + (size_t)t * E_;
    const int* dst = e_dst + (size_t)t * E_;

    // f = feats @ gat_W[t]      (WMMA)
    wmma_gemm<128><<<cdiv(N_A_, 128), 256, 0, stream>>>(
        feats, gat_W + (size_t)t * 128 * 128, nullptr, fbuf,
        N_A_, 128, 128, 0, 0, 0,
        nullptr, nullptr, nullptr, nullptr, nullptr, nullptr, nullptr);

    el_er_kernel<<<cdiv((size_t)N_A_ * 4, 256), 256, 0, stream>>>(
        fbuf, gat_al + t * 128, gat_ar + t * 128, el, er, N_A_);

    fill_u32_kernel<<<cdiv((size_t)N_A_ * 4, 256), 256, 0, stream>>>(
        (unsigned*)mbuf, ENC_NEG_INF, N_A_ * 4);
    hipMemsetAsync(sbuf, 0, (size_t)N_A_ * 4 * 4, stream);
    hipMemsetAsync(hacc, 0, (size_t)N_A_ * 128 * 4, stream);

    gat_edge_score_kernel<<<cdiv((size_t)E_ * 4, 256), 256, 0, stream>>>(
        src, dst, el, er, esc, (unsigned*)mbuf, E_);
    edge_exp_sum_kernel<<<cdiv((size_t)E_ * 4, 256), 256, 0, stream>>>(
        dst, esc, (unsigned*)mbuf, sbuf, E_);
    gat_accum_kernel<<<cdiv((size_t)E_ * 32, 256), 256, 0, stream>>>(
        src, dst, esc, sbuf, fbuf, hacc, E_);
    bias_relu_kernel<<<cdiv((size_t)N_A_ * 128, 256), 256, 0, stream>>>(
        hacc, gat_b + t * 128, hgat, N_A_ * 128);

    // GIN: agg = segsum(h[src]); x = (1+eps)*h + agg
    hipMemsetAsync(fbuf, 0, (size_t)N_A_ * 128 * 4, stream);   // reuse as agg
    scatter_add_kernel<<<cdiv((size_t)E_ * 32, 256), 256, 0, stream>>>(
        src, dst, hgat, fbuf, E_);
    gin_combine_kernel<<<cdiv((size_t)N_A_ * 128, 256), 256, 0, stream>>>(
        hgat, fbuf, gin_ep, t, hacc, N_A_ * 128);

    // x @ gin_W1 + b1          (WMMA)
    wmma_gemm<128><<<cdiv(N_A_, 128), 256, 0, stream>>>(
        hacc, gin_W1 + (size_t)t * 128 * 128, gin_b1 + t * 128, xb,
        N_A_, 128, 128, 0, 0, 0,
        nullptr, nullptr, nullptr, nullptr, nullptr, nullptr, nullptr);

    // BatchNorm stats, then fused BN+ReLU into second GEMM's A-load
    hipMemsetAsync(bns, 0, 256 * 4, stream);
    bn_stats_kernel<<<cdiv(N_A_, 256), 256, 0, stream>>>(xb, bns, N_A_);
    bn_finalize_kernel<<<1, 128, 0, stream>>>(bns, bnm, 1.f / N_A_);

    // relu(BN(x)) @ gin_W2 + b2, relu -> hA[:, t*128 : t*128+128]  (WMMA)
    wmma_gemm<128><<<cdiv(N_A_, 128), 256, 0, stream>>>(
        xb, gin_W2 + (size_t)t * 128 * 128, gin_b2 + t * 128, hA,
        N_A_, 128, 256, t * 128, 2, 1,
        nullptr, nullptr, nullptr, bnm, bnm + 128,
        gin_g + t * 128, gin_be + t * 128);
  }

  // segment max A -> C2 (values >= 0; init 0)
  hipMemsetAsync(hC, 0, (size_t)N_C2_ * 256 * 4, stream);
  segmax_kernel<<<cdiv((size_t)N_A_ * 64, 256), 256, 0, stream>>>(
      hA, g1d, (unsigned*)hC, N_A_);

  // h2 MLP: relu(hC @ h2_W1 + b1) @ h2_W2 + b2   (WMMA x2)
  wmma_gemm<128><<<cdiv(N_C2_, 128), 256, 0, stream>>>(
      hC, h2_W1, h2_b1, hmt, N_C2_, 256, 128, 0, 0, 1,
      nullptr, nullptr, nullptr, nullptr, nullptr, nullptr, nullptr);
  wmma_gemm<32><<<cdiv(N_C2_, 128), 256, 0, stream>>>(
      hmt, h2_W2, h2_b2, hm, N_C2_, 128, 32, 0, 0, 0,
      nullptr, nullptr, nullptr, nullptr, nullptr, nullptr, nullptr);

  // attention MLP layer 1 with fused concat-gather A-load  (WMMA)
  wmma_gemm<128><<<cdiv(EI_, 128), 256, 0, stream>>>(
      nullptr, att_W1, att_b1, tatt, EI_, 128, 128, 0, 1, 1,
      i2s, i2d, pca, nullptr, nullptr, nullptr, nullptr);

  fill_u32_kernel<<<cdiv((size_t)N_C2_ * 4, 256), 256, 0, stream>>>(
      (unsigned*)mC, ENC_NEG_INF, N_C2_ * 4);
  hipMemsetAsync(sC, 0, (size_t)N_C2_ * 4 * 4, stream);
  hipMemsetAsync(hC2, 0, (size_t)N_C2_ * 128 * 4, stream);

  att_edge_score_kernel<<<cdiv((size_t)EI_ * 4, 256), 256, 0, stream>>>(
      tatt, att_W2, att_b2, i2d, asc, (unsigned*)mC, EI_);
  edge_exp_sum_kernel<<<cdiv((size_t)EI_ * 4, 256), 256, 0, stream>>>(
      i2d, asc, (unsigned*)mC, sC, EI_);
  att_accum_kernel<<<cdiv((size_t)EI_ * 32, 256), 256, 0, stream>>>(
      i2s, i2d, asc, sC, hm, hC2, EI_);

  // readout (mean per graph)
  hipMemsetAsync(h1s, 0, (size_t)NB_ * 256 * 4, stream);
  hipMemsetAsync(h2s, 0, (size_t)NB_ * 128 * 4, stream);
  hipMemsetAsync(cA, 0, NB_ * 4, stream);
  hipMemsetAsync(cC, 0, NB_ * 4, stream);
  count_kernel<<<cdiv(N_A_, 256), 256, 0, stream>>>(bA, cA, N_A_);
  count_kernel<<<cdiv(N_C2_, 256), 256, 0, stream>>>(bC, cC, N_C2_);
  readout_sum_kernel<<<cdiv((size_t)N_A_ * 64, 256), 256, 0, stream>>>(
      hA, bA, h1s, N_A_, 256, 0);
  readout_sum_kernel<<<cdiv((size_t)N_C2_ * 32, 256), 256, 0, stream>>>(
      hC2, bC, h2s, N_C2_, 128, 1);
  hcat_kernel<<<cdiv((size_t)NB_ * 384, 256), 256, 0, stream>>>(
      h1s, h2s, cA, cC, hcat);

  // output MLP  (WMMA, single block each)
  wmma_gemm<128><<<1, 256, 0, stream>>>(
      hcat, out_W1, out_b1, f1, NB_, 384, 128, 0, 0, 1,
      nullptr, nullptr, nullptr, nullptr, nullptr, nullptr, nullptr);
  wmma_gemm<128><<<1, 256, 0, stream>>>(
      f1, out_W2, out_b2, f2, NB_, 128, 128, 0, 0, 1,
      nullptr, nullptr, nullptr, nullptr, nullptr, nullptr, nullptr);
  final_out_kernel<<<1, 128, 0, stream>>>(f2, out_W3, out_b3, (float*)d_out);
}
